// TransformerEncoderMOEPLUS_79860621902672
// MI455X (gfx1250) — matrix operand order repaired and checked by reading, not compiled, not run on hardware
//
#include <hip/hip_runtime.h>
#include <hip/hip_bf16.h>

// ---------------------------------------------------------------------------
// TransformerEncoderMOEPLUS for MI455X (gfx1250, wave32, WMMA + TDM).
//  * All GEMMs: v_wmma_f32_16x16x32_bf16, f32 accumulate.
//  * bf16 conversion hoisted out of GEMM: weights pre-converted (and [K,N]
//    weights pre-transposed) to bf16 [N,K]; activations produced in bf16 by
//    the LN / GEMM / attention epilogues.
//  * Tile staging global->LDS uses TENSOR_LOAD_TO_LDS (Tensor Data Mover)
//    with D# descriptors per CDNA5 ISA, DOUBLE-BUFFERED: the DMA for tile
//    k+1 is issued before computing tile k, so HBM/L2 latency overlaps the
//    WMMA burst; s_wait_tensorcnt only at buffer switch (TDM loads from one
//    wave complete in order). Falls back to uint4 copies without the builtin.
// ---------------------------------------------------------------------------

typedef __attribute__((ext_vector_type(16))) __bf16   bf16x16;
typedef __attribute__((ext_vector_type(8)))  float    floatx8;
typedef __attribute__((ext_vector_type(4)))  unsigned u32x4;
typedef __attribute__((ext_vector_type(8)))  int      i32x8;
typedef __attribute__((ext_vector_type(4)))  int      i32x4;

#define DEV static __device__ __forceinline__

#if __has_builtin(__builtin_amdgcn_tensor_load_to_lds) && \
    __has_builtin(__builtin_amdgcn_s_wait_tensorcnt)
#define HAVE_TDM 1
#else
#define HAVE_TDM 0
#endif

constexpr int S_    = 1024;
constexpr int B_    = 2;
constexpr int D_    = 1024;
constexpr int NH_   = 16;
constexpr int HD_   = 64;
constexpr int NTOK  = S_ * B_;     // 2048 token rows
constexpr int DFF_  = 4 * D_;      // 4096
constexpr int E_    = 4;
constexpr float LN_EPS_ = 1e-5f;

union Frag16 {                      // 32 bytes: one bf16 WMMA A/B fragment
  bf16x16 v;
  uint4   q[2];
};

DEV __bf16 f2bf(float x) { return (__bf16)x; }

DEV floatx8 wmma_bf16(bf16x16 a, bf16x16 b, floatx8 c) {
  return __builtin_amdgcn_wmma_f32_16x16x32_bf16(false, a, false, b,
                                                 (short)0, c, false, false);
}

DEV floatx8 zero8() {
  floatx8 z = {0.f, 0.f, 0.f, 0.f, 0.f, 0.f, 0.f, 0.f};
  return z;
}

#if HAVE_TDM
// Generic LDS byte offset of a __shared__ object (LDS aperture: addr[31:0]).
DEV unsigned lds_off(const void* p) { return (unsigned)(unsigned long long)p; }

// 2-D bf16 tile DMA: rows x cols elements, row stride strideElems, into LDS
// with a 16B pad appended per row (pad_amount code 3 = 4 DWORDs).
// padIntervalCode: 3 -> pad every 64B row (cols=32), 4 -> every 128B (cols=64).
DEV void tdm_tile_load_bf16(const void* gsrc, unsigned ldsByteOff,
                            unsigned rows, unsigned cols,
                            unsigned strideElems, unsigned padIntervalCode) {
  unsigned long long ga = (unsigned long long)gsrc;
  u32x4 g0;
  g0[0] = 1u;                                  // count=1 (valid descriptor)
  g0[1] = ldsByteOff;                          // lds_addr
  g0[2] = (unsigned)ga;                        // global_addr[31:0]
  g0[3] = (unsigned)(ga >> 32) | (2u << 30);   // global_addr[56:32] | type=2
  i32x8 g1;
  g1[0] = (int)((1u << 16)                      // data_size = 2 bytes
              | (1u << 20)                      // pad_enable
              | (padIntervalCode << 22)         // pad_interval
              | (3u << 25));                    // pad_amount = 4 DWORDs (16B)
  g1[1] = (int)((cols & 0xFFFFu) << 16);        // tensor_dim0 = cols
  g1[2] = (int)((rows & 0xFFFFu) << 16);        // tensor_dim1 = rows
  g1[3] = (int)((cols & 0xFFFFu) << 16);        // tile_dim0 = cols
  g1[4] = (int)(rows & 0xFFFFu);                // tile_dim1 = rows
  g1[5] = (int)strideElems;                     // tensor_dim0_stride[31:0]
  g1[6] = 0;
  g1[7] = 0;
  i32x4 z4 = {0, 0, 0, 0};
#if __has_include(<hip/amd_detail/amd_gfx1250_TDM.h>)
  i32x8 z8 = {0, 0, 0, 0, 0, 0, 0, 0};          // clang-23 / therock: 6 args
  __builtin_amdgcn_tensor_load_to_lds(g0, g1, z4, z4, z8, 0);
#else
  __builtin_amdgcn_tensor_load_to_lds(g0, g1, z4, z4, 0);  // ROCm 7.2: 5 args
#endif
}
#endif  // HAVE_TDM

// ---------------------------------------------------------------------------
// x = sqrt(D) * x_in + sinusoidal_pos_emb(s)
// ---------------------------------------------------------------------------
__global__ __launch_bounds__(256) void embed_kernel(const float* __restrict__ xin,
                                                    float* __restrict__ X) {
  int idx = blockIdx.x * 256 + threadIdx.x;
  int d   = idx & (D_ - 1);
  int row = idx >> 10;
  int s   = row / B_;
  const int half = D_ / 2;
  int i = (d < half) ? d : d - half;
  float freq = __expf((float)i * (-9.210340371976184f / 511.0f));
  float ang  = (float)s * freq;
  float pe   = (d < half) ? __sinf(ang) : __cosf(ang);
  X[idx] = 32.0f * xin[idx] + pe;
}

// ---------------------------------------------------------------------------
// LayerNorm (D=1024), one 256-thread block per row; optional f32 and bf16 out.
// ---------------------------------------------------------------------------
__global__ __launch_bounds__(256) void layernorm_kernel(const float* __restrict__ x,
                                                        const float* __restrict__ g,
                                                        const float* __restrict__ bp,
                                                        float* __restrict__ y,
                                                        __bf16* __restrict__ yb) {
  __shared__ float s1[8], s2[8];
  int row = blockIdx.x, tid = threadIdx.x;
  const float* xr = x + (size_t)row * D_;
  float v[4]; float sum = 0.f;
  #pragma unroll
  for (int i = 0; i < 4; i++) { v[i] = xr[tid + i * 256]; sum += v[i]; }
  #pragma unroll
  for (int m = 1; m < 32; m <<= 1) sum += __shfl_xor(sum, m, 32);
  if ((tid & 31) == 0) s1[tid >> 5] = sum;
  __syncthreads();
  float tot = 0.f;
  #pragma unroll
  for (int i = 0; i < 8; i++) tot += s1[i];
  float mu = tot * (1.0f / D_);
  float vs = 0.f;
  #pragma unroll
  for (int i = 0; i < 4; i++) { float dl = v[i] - mu; vs += dl * dl; }
  #pragma unroll
  for (int m = 1; m < 32; m <<= 1) vs += __shfl_xor(vs, m, 32);
  if ((tid & 31) == 0) s2[tid >> 5] = vs;
  __syncthreads();
  float vtot = 0.f;
  #pragma unroll
  for (int i = 0; i < 8; i++) vtot += s2[i];
  float rs = rsqrtf(vtot * (1.0f / D_) + LN_EPS_);
  #pragma unroll
  for (int i = 0; i < 4; i++) {
    int d = tid + i * 256;
    float o = (v[i] - mu) * rs * g[d] + bp[d];
    if (y)  y[(size_t)row * D_ + d] = o;
    if (yb) yb[(size_t)row * D_ + d] = f2bf(o);
  }
}

// ---------------------------------------------------------------------------
// f32 -> bf16 elementwise (for [N,K]-stored weights; n multiple of 4)
// ---------------------------------------------------------------------------
__global__ __launch_bounds__(256) void cvt_bf16_kernel(const float* __restrict__ src,
                                                       __bf16* __restrict__ dst,
                                                       int n) {
  int i = (blockIdx.x * 256 + threadIdx.x) * 4;
  if (i >= n) return;
  float4 f = *(const float4*)(src + i);
  __bf16 o[4] = {f2bf(f.x), f2bf(f.y), f2bf(f.z), f2bf(f.w)};
  *(uint2*)(dst + i) = *(const uint2*)o;
}

// ---------------------------------------------------------------------------
// W [K,N] f32  ->  Wt [N,K] bf16 (32x32 LDS tiles, coalesced both sides)
// ---------------------------------------------------------------------------
__global__ __launch_bounds__(256) void transpose_cvt_kernel(const float* __restrict__ W,
                                                            __bf16* __restrict__ Wt,
                                                            int Kdim, int N) {
  __shared__ __bf16 t[32][33];
  int k0 = blockIdx.y * 32, n0 = blockIdx.x * 32;
  int tx = threadIdx.x & 31, ty = threadIdx.x >> 5;   // ty 0..7
  #pragma unroll
  for (int i = 0; i < 4; i++) {
    int k = ty + i * 8;
    t[k][tx] = f2bf(W[(size_t)(k0 + k) * N + n0 + tx]);
  }
  __syncthreads();
  #pragma unroll
  for (int i = 0; i < 4; i++) {
    int nn = ty + i * 8;
    Wt[(size_t)(n0 + nn) * Kdim + k0 + tx] = t[tx][nn];
  }
}

// ---------------------------------------------------------------------------
// bf16 WMMA GEMM, A bf16 [M,K], Wt bf16 [N,K] (both K-contiguous):
//   out[m,n] = (residual?) + (rowScale[m]?) * outScale*(A·Wt^T + bias[n])
// Tiles 128x128x32, 8 waves (4Mx2N), wave = 2x4 WMMA frags. Tile staging via
// double-buffered TDM: DMA of tile k+1 overlaps the WMMA burst of tile k.
// ---------------------------------------------------------------------------
constexpr int BM = 128, BN = 128, BK = 32;
constexpr int LDT = BK + 8;   // padded pitch in elements (80B rows, 16B pad)

__global__ __launch_bounds__(256) void gemm_bf16_wmma(
    const __bf16* __restrict__ A, const __bf16* __restrict__ Wt,
    const float* __restrict__ bias, const float* __restrict__ residual,
    const float* __restrict__ rowScale, int rsStride,
    float* __restrict__ Cf, __bf16* __restrict__ Cb,
    int N, int Kd, int doRelu, float outScale) {
  __shared__ __bf16 As[2][BM * LDT];
  __shared__ __bf16 Bs[2][BN * LDT];

  const int tid  = threadIdx.x;
  const int lane = tid & 31;
  const int wave = tid >> 5;
  const int half = lane >> 4, r = lane & 15;
  const int m0 = blockIdx.y * BM, n0 = blockIdx.x * BN;
  const int mwb = (wave & 3) * 32;
  const int nwb = (wave >> 2) * 64;
  const int nsteps = Kd / BK;

  floatx8 acc[2][4];
  #pragma unroll
  for (int mi = 0; mi < 2; mi++)
    #pragma unroll
    for (int ni = 0; ni < 4; ni++) acc[mi][ni] = zero8();

#if HAVE_TDM
  if (tid < 32) {   // prologue: wave 0 issues DMAs for tile 0 (TDM ignores EXEC)
    tdm_tile_load_bf16(A + (size_t)m0 * Kd, lds_off(As[0]), BM, BK,
                       (unsigned)Kd, 3u);
    tdm_tile_load_bf16(Wt + (size_t)n0 * Kd, lds_off(Bs[0]), BN, BK,
                       (unsigned)Kd, 3u);
  }
#endif

  for (int s = 0; s < nsteps; s++) {
    const int cur = s & 1;
    const int k0 = s * BK;
#if HAVE_TDM
    if (tid < 32) __builtin_amdgcn_s_wait_tensorcnt(0);  // buffer 'cur' ready
    __syncthreads();
    if (tid < 32 && (s + 1) < nsteps) {   // prefetch tile s+1 into other buffer
      tdm_tile_load_bf16(A + (size_t)m0 * Kd + k0 + BK, lds_off(As[cur ^ 1]),
                         BM, BK, (unsigned)Kd, 3u);
      tdm_tile_load_bf16(Wt + (size_t)n0 * Kd + k0 + BK, lds_off(Bs[cur ^ 1]),
                         BN, BK, (unsigned)Kd, 3u);
    }
#else
    #pragma unroll
    for (int it = 0; it < 2; it++) {   // 512 16B chunks / 256 threads
      int idx = it * 256 + tid;
      int row = idx >> 2, c = idx & 3;
      *(uint4*)&As[cur][row * LDT + c * 8] =
          *(const uint4*)(A + (size_t)(m0 + row) * Kd + k0 + c * 8);
      *(uint4*)&Bs[cur][row * LDT + c * 8] =
          *(const uint4*)(Wt + (size_t)(n0 + row) * Kd + k0 + c * 8);
    }
    __syncthreads();
#endif

    Frag16 a[2], bw[4];
    #pragma unroll
    for (int mi = 0; mi < 2; mi++) {
      const __bf16* rowp = &As[cur][(mwb + mi * 16 + r) * LDT];
      a[mi].q[0] = *(const uint4*)(rowp + half * 16);
      a[mi].q[1] = *(const uint4*)(rowp + half * 16 + 8);
    }
    #pragma unroll
    for (int ni = 0; ni < 4; ni++) {
      const __bf16* colp = &Bs[cur][(nwb + ni * 16 + r) * LDT + half * 16];
      bw[ni].q[0] = ((const uint4*)colp)[0];
      bw[ni].q[1] = ((const uint4*)colp)[1];
    }
    #pragma unroll
    for (int mi = 0; mi < 2; mi++)
      #pragma unroll
      for (int ni = 0; ni < 4; ni++)
        acc[mi][ni] = wmma_bf16(a[mi].v, bw[ni].v, acc[mi][ni]);
    __syncthreads();
  }

  #pragma unroll
  for (int mi = 0; mi < 2; mi++)
    #pragma unroll
    for (int ni = 0; ni < 4; ni++) {
      int n = n0 + nwb + ni * 16 + r;
      float bv = bias ? bias[n] : 0.f;
      #pragma unroll
      for (int v = 0; v < 8; v++) {
        int m = m0 + mwb + mi * 16 + half * 8 + v;
        float val = acc[mi][ni][v] + bv;
        if (doRelu)   val = fmaxf(val, 0.f);
        val *= outScale;
        if (rowScale) val *= rowScale[(size_t)m * rsStride];
        if (residual) val += residual[(size_t)m * N + n];
        if (Cf) Cf[(size_t)m * N + n] = val;
        else    Cb[(size_t)m * N + n] = f2bf(val);
      }
    }
}

// ---------------------------------------------------------------------------
// Flash attention, HD=64, bf16 Q/K/V (q pre-scaled in the Q-projection GEMM).
// Block = one (b,h) x 64 queries (4 waves x 16 queries). K/V 32x64 tiles are
// staged once per block with double-buffered TDM (DMA of chunk t+1 overlaps
// softmax+PV of chunk t). Online softmax via 16-lane butterflies; P is
// re-laid-out C->A through per-wave LDS.
// ---------------------------------------------------------------------------
__global__ __launch_bounds__(128) void flash_attn_kernel(
    const __bf16* __restrict__ Q, const __bf16* __restrict__ Kg,
    const __bf16* __restrict__ V, __bf16* __restrict__ AO) {
  __shared__ __bf16 Kt[2][32 * 72];      // [t][d], 128B rows + 16B pad
  __shared__ __bf16 Vt[2][32 * 72];
  __shared__ __bf16 Pst[4][16 * 40];     // per-wave 16x32 P tile

  const int tid  = threadIdx.x;
  const int lane = tid & 31, wave = tid >> 5;
  const int half = lane >> 4, r = lane & 15;
  const int bid = blockIdx.x;
  const int bh  = bid >> 4;              // b*NH + h
  const int qb4 = bid & 15;
  const int b = bh / NH_, h = bh % NH_;
  const int qbase = (qb4 * 4 + wave) * 16;

  Frag16 qa[2];
  {
    const __bf16* qrow = Q + ((size_t)(qbase + r) * B_ + b) * D_ + h * HD_;
    qa[0].q[0] = *(const uint4*)(qrow + half * 8);
    qa[0].q[1] = *(const uint4*)(qrow + 16 + half * 8);
    qa[1].q[0] = *(const uint4*)(qrow + 32 + half * 8);
    qa[1].q[1] = *(const uint4*)(qrow + 48 + half * 8);
  }

  floatx8 Oacc[4];
  #pragma unroll
  for (int ni = 0; ni < 4; ni++) Oacc[ni] = zero8();
  float rmax[8], rsum[8];
  #pragma unroll
  for (int v = 0; v < 8; v++) { rmax[v] = -INFINITY; rsum[v] = 0.f; }

  __bf16* Pw = &Pst[wave][0];
  const size_t kvBase = ((size_t)0 * B_ + b) * D_ + h * HD_;
  const unsigned kvStride = (unsigned)(B_ * D_);   // elements between t rows

#if HAVE_TDM
  if (tid < 32) {   // prologue: stage chunk 0
    tdm_tile_load_bf16(Kg + kvBase, lds_off(Kt[0]), 32, 64, kvStride, 4u);
    tdm_tile_load_bf16(V + kvBase, lds_off(Vt[0]), 32, 64, kvStride, 4u);
  }
#endif

  for (int s = 0; s < S_ / 32; s++) {
    const int cur = s & 1;
    const int t0 = s * 32;
#if HAVE_TDM
    if (tid < 32) __builtin_amdgcn_s_wait_tensorcnt(0);
    __syncthreads();
    if (tid < 32 && (s + 1) < S_ / 32) {   // prefetch chunk s+1
      size_t goff = ((size_t)(t0 + 32) * B_ + b) * D_ + h * HD_;
      tdm_tile_load_bf16(Kg + goff, lds_off(Kt[cur ^ 1]), 32, 64, kvStride, 4u);
      tdm_tile_load_bf16(V + goff, lds_off(Vt[cur ^ 1]), 32, 64, kvStride, 4u);
    }
#else
    #pragma unroll
    for (int it = 0; it < 2; it++) {       // 256 16B chunks per tile, 128 thr
      int idx = it * 128 + tid;
      int rowt = idx >> 3, c = idx & 7;
      size_t goff = ((size_t)(t0 + rowt) * B_ + b) * D_ + h * HD_ + c * 8;
      *(uint4*)&Kt[cur][rowt * 72 + c * 8] = *(const uint4*)(Kg + goff);
      *(uint4*)&Vt[cur][rowt * 72 + c * 8] = *(const uint4*)(V + goff);
    }
    __syncthreads();
#endif

    // ---- scores (QK^T) ----
    floatx8 sc[2];
    #pragma unroll
    for (int tc = 0; tc < 2; tc++) {
      const __bf16* krow = &Kt[cur][(tc * 16 + r) * 72];
      Frag16 kb0, kb1;
      kb0.q[0] = *(const uint4*)(krow + half * 8);
      kb0.q[1] = *(const uint4*)(krow + 16 + half * 8);
      kb1.q[0] = *(const uint4*)(krow + 32 + half * 8);
      kb1.q[1] = *(const uint4*)(krow + 48 + half * 8);
      sc[tc] = wmma_bf16(qa[0].v, kb0.v, zero8());
      sc[tc] = wmma_bf16(qa[1].v, kb1.v, sc[tc]);
    }

    // ---- online softmax over this 32-wide t chunk ----
    float corr[8];
    #pragma unroll
    for (int v = 0; v < 8; v++) {
      float mx = fmaxf(sc[0][v], sc[1][v]);
      #pragma unroll
      for (int m = 1; m < 16; m <<= 1) mx = fmaxf(mx, __shfl_xor(mx, m, 32));
      float nmax = fmaxf(rmax[v], mx);
      corr[v] = __expf(rmax[v] - nmax);
      rmax[v] = nmax;
    }
    #pragma unroll
    for (int v = 0; v < 8; v++) {
      float p0 = __expf(sc[0][v] - rmax[v]);
      float p1 = __expf(sc[1][v] - rmax[v]);
      sc[0][v] = p0; sc[1][v] = p1;
      float sm = p0 + p1;
      #pragma unroll
      for (int m = 1; m < 16; m <<= 1) sm += __shfl_xor(sm, m, 32);
      rsum[v] = rsum[v] * corr[v] + sm;
      #pragma unroll
      for (int ni = 0; ni < 4; ni++) Oacc[ni][v] *= corr[v];
    }

    // ---- P: C layout -> per-wave LDS -> A layout ----
    #pragma unroll
    for (int v = 0; v < 8; v++) {
      Pw[(half * 8 + v) * 40 + r]      = f2bf(sc[0][v]);
      Pw[(half * 8 + v) * 40 + 16 + r] = f2bf(sc[1][v]);
    }
    __syncthreads();
    Frag16 pa;
    pa.q[0] = *(const uint4*)&Pw[r * 40 + half * 16];
    pa.q[1] = *(const uint4*)&Pw[r * 40 + half * 16 + 8];

    // ---- O += P x V ----
    #pragma unroll
    for (int ni = 0; ni < 4; ni++) {
      Frag16 vb;
      #pragma unroll
      for (int e = 0; e < 16; e++)
        vb.v[e] = Vt[cur][(half * 16 + e) * 72 + ni * 16 + r];
      Oacc[ni] = wmma_bf16(pa.v, vb.v, Oacc[ni]);
    }
    __syncthreads();   // all reads of buffer 'cur' + P done
  }

  #pragma unroll
  for (int ni = 0; ni < 4; ni++)
    #pragma unroll
    for (int v = 0; v < 8; v++) {
      int s = qbase + half * 8 + v;
      int d = h * HD_ + ni * 16 + r;
      AO[((size_t)s * B_ + b) * D_ + d] = f2bf(Oacc[ni][v] / rsum[v]);
    }
}

// ---------------------------------------------------------------------------
// NoisyTopK router (exact-zero gates for unselected experts)
// ---------------------------------------------------------------------------
__global__ __launch_bounds__(256) void moe_router_kernel(
    const float* __restrict__ Y,
    const float* __restrict__ Wr, const float* __restrict__ br,
    const float* __restrict__ Wn, const float* __restrict__ bn,
    const float* __restrict__ eps, float* __restrict__ gates) {
  int wave = threadIdx.x >> 5, lane = threadIdx.x & 31;
  int tok = blockIdx.x * 8 + wave;
  const float* y = Y + (size_t)tok * D_;
  float accr[E_] = {0, 0, 0, 0}, accn[E_] = {0, 0, 0, 0};
  for (int i = 0; i < 32; i++) {
    int d = lane * 32 + i;
    float yv = y[d];
    #pragma unroll
    for (int e = 0; e < E_; e++) {
      accr[e] += yv * Wr[e * D_ + d];
      accn[e] += yv * Wn[e * D_ + d];
    }
  }
  #pragma unroll
  for (int e = 0; e < E_; e++)
    #pragma unroll
    for (int m = 1; m < 32; m <<= 1) {
      accr[e] += __shfl_xor(accr[e], m, 32);
      accn[e] += __shfl_xor(accn[e], m, 32);
    }
  if (lane == 0) {
    float noisy[E_];
    #pragma unroll
    for (int e = 0; e < E_; e++) {
      float nl = accn[e] + bn[e];
      float sp = (nl > 0.f) ? nl + log1pf(__expf(-nl)) : log1pf(__expf(nl));
      noisy[e] = accr[e] + br[e] + eps[(size_t)tok * E_ + e] * sp;
    }
    int i0 = 0;
    #pragma unroll
    for (int e = 1; e < E_; e++) if (noisy[e] > noisy[i0]) i0 = e;
    int i1 = -1;
    #pragma unroll
    for (int e = 0; e < E_; e++)
      if (e != i0 && (i1 < 0 || noisy[e] > noisy[i1])) i1 = e;
    float mx = noisy[i0];
    float e0 = __expf(noisy[i0] - mx), e1 = __expf(noisy[i1] - mx);
    float inv = 1.0f / (e0 + e1);
    #pragma unroll
    for (int e = 0; e < E_; e++)
      gates[(size_t)tok * E_ + e] = (e == i0) ? e0 * inv
                                  : (e == i1) ? e1 * inv : 0.f;
  }
}

// ---------------------------------------------------------------------------
// Host orchestration (param order: x_in, noise_eps, 2 tf layers x16,
// 2 moe layers x20, ln_g, ln_b)
// ---------------------------------------------------------------------------
extern "C" void kernel_launch(void* const* d_in, const int* in_sizes, int n_in,
                              void* d_out, int out_size, void* d_ws, size_t ws_size,
                              hipStream_t stream) {
  (void)in_sizes; (void)out_size; (void)ws_size;
  const float* x_in      = (const float*)d_in[0];
  const float* noise_eps = (const float*)d_in[1];
  auto P = [&](int i) { return (const float*)d_in[i]; };

  char* ws = (char*)d_ws;
  const size_t MB = 1u << 20;
  float*  X   = (float*)(ws + 0 * MB);     // 8 MB  f32 residual stream
  float*  Y   = (float*)(ws + 8 * MB);     // 8 MB  f32 LN out (router input)
  __bf16* Yb  = (__bf16*)(ws + 16 * MB);   // 4 MB  bf16 LN out (GEMM A)
  __bf16* Qb  = (__bf16*)(ws + 20 * MB);   // 4 MB
  __bf16* Kb  = (__bf16*)(ws + 24 * MB);   // 4 MB
  __bf16* Vb  = (__bf16*)(ws + 28 * MB);   // 4 MB
  __bf16* AOb = (__bf16*)(ws + 32 * MB);   // 4 MB
  __bf16* Hb  = (__bf16*)(ws + 36 * MB);   // 16 MB (NTOK x DFF)
  __bf16* WB  = (__bf16*)(ws + 52 * MB);   // 8 MB  bf16 weight scratch [N,K]
  float*  GT  = (float*)(ws + 60 * MB);    // gates NTOK x E

  embed_kernel<<<(NTOK * D_) / 256, 256, 0, stream>>>(x_in, X);

  dim3 gProj(D_ / BN, NTOK / BM);
  dim3 gFF1(DFF_ / BN, NTOK / BM);
  dim3 gFF2(D_ / BN, NTOK / BM);

  auto cvtNK = [&](const float* W, int elems) {           // W already [N,K]
    cvt_bf16_kernel<<<elems / 4 / 256, 256, 0, stream>>>(W, WB, elems);
  };
  auto cvtKN = [&](const float* W, int Kdim, int N) {     // W [K,N] -> [N,K]
    transpose_cvt_kernel<<<dim3(N / 32, Kdim / 32), 256, 0, stream>>>(W, WB, Kdim, N);
  };
  auto gemm = [&](const __bf16* A, const float* bias, const float* res,
                  const float* rs, int rss, float* Cf, __bf16* Cb,
                  int N, int Kd, int relu, float oscale, dim3 grid) {
    gemm_bf16_wmma<<<grid, 256, 0, stream>>>(A, WB, bias, res, rs, rss,
                                             Cf, Cb, N, Kd, relu, oscale);
  };

  for (int L = 0; L < 4; L++) {
    const bool moe = (L >= 2);
    const int base = moe ? (2 + 2 * 16 + (L - 2) * 20) : (2 + L * 16);
    const float *Wq = P(base + 0),  *bq = P(base + 1);
    const float *Wk = P(base + 2),  *bk = P(base + 3);
    const float *Wv = P(base + 4),  *bv = P(base + 5);
    const float *Wo = P(base + 6),  *bo = P(base + 7);
    const float *ln0g = P(base + 8),  *ln0b = P(base + 9);
    const float *ln1g = P(base + 10), *ln1b = P(base + 11);

    // ---- pre-LN MHA + residual ----
    layernorm_kernel<<<NTOK, 256, 0, stream>>>(X, ln0g, ln0b, nullptr, Yb);
    cvtNK(Wq, D_ * D_);
    gemm(Yb, bq, nullptr, nullptr, 0, nullptr, Qb, D_, D_, 0, 0.125f, gProj);
    cvtNK(Wk, D_ * D_);
    gemm(Yb, bk, nullptr, nullptr, 0, nullptr, Kb, D_, D_, 0, 1.0f, gProj);
    cvtNK(Wv, D_ * D_);
    gemm(Yb, bv, nullptr, nullptr, 0, nullptr, Vb, D_, D_, 0, 1.0f, gProj);
    flash_attn_kernel<<<(S_ / 64) * B_ * NH_, 128, 0, stream>>>(Qb, Kb, Vb, AOb);
    cvtNK(Wo, D_ * D_);
    gemm(AOb, bo, X, nullptr, 0, X, nullptr, D_, D_, 0, 1.0f, gProj);

    // ---- pre-LN FFN / MoE + residual ----
    layernorm_kernel<<<NTOK, 256, 0, stream>>>(X, ln1g, ln1b,
                                               moe ? Y : nullptr, Yb);
    if (!moe) {
      const float *W1 = P(base + 12), *b1 = P(base + 13);
      const float *W2 = P(base + 14), *b2 = P(base + 15);
      cvtKN(W1, D_, DFF_);
      gemm(Yb, b1, nullptr, nullptr, 0, nullptr, Hb, DFF_, D_, 1, 1.0f, gFF1);
      cvtKN(W2, DFF_, D_);
      gemm(Hb, b2, X, nullptr, 0, X, nullptr, D_, DFF_, 0, 1.0f, gFF2);
    } else {
      const int j = L - 2;
      const float *Wr = P(base + 12), *br = P(base + 13);
      const float *Wn = P(base + 14), *bn = P(base + 15);
      const float *W1 = P(base + 16), *b1 = P(base + 17);
      const float *W2 = P(base + 18), *b2 = P(base + 19);
      moe_router_kernel<<<NTOK / 8, 256, 0, stream>>>(
          Y, Wr, br, Wn, bn, noise_eps + (size_t)j * NTOK * E_, GT);
      for (int e = 0; e < E_; e++) {
        cvtKN(W1 + (size_t)e * D_ * DFF_, D_, DFF_);
        gemm(Yb, b1 + (size_t)e * DFF_, nullptr, nullptr, 0,
             nullptr, Hb, DFF_, D_, 1, 1.0f, gFF1);
        cvtKN(W2 + (size_t)e * DFF_ * D_, DFF_, D_);
        gemm(Hb, b2 + (size_t)e * D_, X, GT + e, E_,
             X, nullptr, D_, DFF_, 0, 1.0f, gFF2);
      }
    }
  }

  layernorm_kernel<<<NTOK, 256, 0, stream>>>(
      X, (const float*)d_in[n_in - 2], (const float*)d_in[n_in - 1],
      (float*)d_out, nullptr);
}
